// SupervisedECGGraph_23029614641393
// MI455X (gfx1250) — compile-verified, matrix operand order checked by reading, not run on hardware
//
#include <hip/hip_runtime.h>
#include <hip/hip_bf16.h>

typedef __attribute__((ext_vector_type(16))) _Float16 v16h;
typedef __attribute__((ext_vector_type(8)))  float    v8f;

#define BATCH  16
#define NB     1024
#define BN     (BATCH * NB)
#define DIM    64
#define LPAD   404   // 400 + 2 pad each side
#define H1W    208   // padded conv1 row (204 used)
#define KK2    160   // conv2 GEMM K (= 32 cin * 5 taps)
#define NPOS   112   // conv2 GEMM N (100 positions padded to 7 tiles)

// ---------------- WMMA fragment helper (CDNA5 16x16x32 f16 layout) ----------------
// Loads a 16x32 (MxK) fragment from a row-major source with leading dim `ld` (halves).
// Lanes 0-15 = rows 0-15 with K-halves {0-7,16-23}; lanes 16-31 same rows, {8-15,24-31}.
// Per-lane reads are two contiguous 16B runs -> vectorizable to b128 loads.
// The same layout is used to feed the B operand with a (transposed-in-memory) source.
__device__ __forceinline__ v16h frag_rowmajor(const _Float16* src, int ld, int lane) {
    int r  = lane & 15;
    int hb = (lane >> 4) << 3;              // K half-split: 0 or 8
    const _Float16* p = src + (size_t)r * ld;
    v16h a;
#pragma unroll
    for (int e = 0; e < 8; ++e) { a[e] = p[hb + e]; a[8 + e] = p[16 + hb + e]; }
    return a;
}

// ---------------- small converts ----------------
__global__ void convert_w2_kernel(const float* w2, _Float16* w2r) {
    int i = blockIdx.x * blockDim.x + threadIdx.x;
    if (i < 64 * KK2) w2r[i] = (_Float16)w2[i];   // (cout, cin*5) already contiguous
}

// X (BN,64) f32 -> XT (BATCH, 64, NB) f16 ; output-major so stores are coalesced
__global__ void transpose16_kernel(const float* __restrict__ X, _Float16* __restrict__ XT) {
    int o = blockIdx.x * blockDim.x + threadIdx.x;     // ((b*64+d)*NB + n)
    if (o < BN * DIM) {
        int n  = o & (NB - 1);
        int bd = o >> 10;
        int d  = bd & (DIM - 1);
        int b  = bd >> 6;
        XT[o] = (_Float16)X[((size_t)b * NB + n) * DIM + d];
    }
}

// ---------------- 1) Beat encoder: conv1 (VALU) + conv2 (WMMA im2col GEMM) + mean + fc ----
__global__ __launch_bounds__(256) void encoder_kernel(
    const float* __restrict__ beats, const float* __restrict__ rr,
    const float* __restrict__ w1, const float* __restrict__ b1,
    const float* __restrict__ b2c, const _Float16* __restrict__ w2r,
    const float* __restrict__ fcw, const float* __restrict__ fcb,
    float* __restrict__ X)
{
    __shared__ float    ipad[LPAD];
    __shared__ _Float16 h1[32][H1W];
    __shared__ _Float16 h2[NPOS][KK2];     // im2col^T: [position][cin*5+k]  (35 KB)
    __shared__ float    chsum[64];
    int t    = threadIdx.x;
    int beat = blockIdx.x;                          // 0..BN-1
    const float* bp = beats + (size_t)beat * 400;

    for (int i = t; i < LPAD; i += 256) {
        int x = i - 2;
        ipad[i] = (x >= 0 && x < 400) ? bp[x] : 0.0f;
    }
    if (t < 64) chsum[t] = 0.0f;
    for (int i = t; i < 32 * H1W; i += 256) (&h1[0][0])[i] = (_Float16)0.0f;
    __syncthreads();

    // conv1: 32 ch x 200 pos, stride2 pad2, relu -> f16 LDS (offset +2 for pad)
    for (int i = t; i < 32 * 200; i += 256) {
        int c = i / 200, p = i % 200;
        float acc = b1[c];
#pragma unroll
        for (int k = 0; k < 5; ++k) acc += w1[c * 5 + k] * ipad[2 * p + k];
        h1[c][p + 2] = (_Float16)(acc > 0.0f ? acc : 0.0f);
    }
    __syncthreads();

    // im2col (transposed): h2[p][ci*5+k] = h1[ci][2p+k] ; p>=100 zero-padded.
    // One-time scatter so the WMMA K-loop below reads contiguous 16B runs.
    for (int i = t; i < NPOS * KK2; i += 256) {
        int p = i / KK2, kk = i % KK2;
        int ci = kk / 5, k = kk % 5;
        (&h2[0][0])[i] = (p < 100) ? h1[ci][2 * p + k] : (_Float16)0.0f;
    }
    __syncthreads();

    // conv2 GEMM: out[co][p] = b2[co] + sum_kk w2r[co][kk] * h2[p][kk]
    // M=64 (4 tiles) x N=112 (7 tiles) x K=160 (5 WMMA steps)
    int wave = t >> 5, lane = t & 31;
    for (int tile = wave; tile < 28; tile += 8) {
        int cotile = tile & 3;
        int ptile  = tile >> 2;
        v8f acc = {};
#pragma unroll
        for (int kt = 0; kt < 5; ++kt) {
            v16h af = frag_rowmajor(w2r + (size_t)(cotile * 16) * KK2 + kt * 32, KK2, lane);
            v16h bf = frag_rowmajor(&h2[ptile * 16][kt * 32], KK2, lane);
            acc = __builtin_amdgcn_wmma_f32_16x16x32_f16(false, af, false, bf,
                                                         (short)0, acc, false, false);
        }
        int col   = ptile * 16 + (lane & 15);
        int rbase = cotile * 16 + ((lane >> 4) << 3);
        if (col < 100) {
#pragma unroll
            for (int v = 0; v < 8; ++v) {
                float val = acc[v] + b2c[rbase + v];
                val = val > 0.0f ? val : 0.0f;      // relu then mean
                atomicAdd(&chsum[rbase + v], val);
            }
        }
    }
    __syncthreads();

    // fc: 66 -> 64  (mean = chsum/100, concat rr)
    if (t < 64) {
        float rr0 = rr[(size_t)beat * 2 + 0], rr1 = rr[(size_t)beat * 2 + 1];
        const float* wrow = fcw + t * 66;
        float acc = fcb[t];
#pragma unroll 8
        for (int i = 0; i < 64; ++i) acc += (chsum[i] * 0.01f) * wrow[i];
        acc += rr0 * wrow[64] + rr1 * wrow[65];
        X[(size_t)beat * DIM + t] = acc;
    }
}

// ---------------- 2) row normalize -> f16 ----------------
__global__ __launch_bounds__(64) void rownorm_kernel(const float* __restrict__ X,
                                                     _Float16* __restrict__ Xn) {
    int row = blockIdx.x, t = threadIdx.x;
    __shared__ float red[64];
    float v = X[(size_t)row * DIM + t];
    red[t] = v * v;  __syncthreads();
    for (int s = 32; s > 0; s >>= 1) { if (t < s) red[t] += red[t + s]; __syncthreads(); }
    float inv = 1.0f / (sqrtf(red[0]) + 1e-8f);
    Xn[(size_t)row * DIM + t] = (_Float16)(v * inv);
}

// ---------------- 3) A = relu(Xn Xn^T), diag=0 : 8 waves/block, 1 tile/wave ----------------
__global__ __launch_bounds__(256) void adjacency_kernel(const _Float16* __restrict__ Xn,
                                                        _Float16* __restrict__ A) {
    int ti = blockIdx.x;
    int tj = blockIdx.y * 8 + threadIdx.y;
    int b  = blockIdx.z;
    int lane = threadIdx.x;
    const _Float16* xb = Xn + (size_t)b * NB * DIM;
    v8f c = {};
#pragma unroll
    for (int kt = 0; kt < 2; ++kt) {
        v16h af = frag_rowmajor(xb + (size_t)ti * 16 * DIM + kt * 32, DIM, lane);
        v16h bf = frag_rowmajor(xb + (size_t)tj * 16 * DIM + kt * 32, DIM, lane); // B = Xn^T
        c = __builtin_amdgcn_wmma_f32_16x16x32_f16(false, af, false, bf, (short)0, c, false, false);
    }
    _Float16* Ab = A + (size_t)b * NB * NB;
    int col   = tj * 16 + (lane & 15);
    int rbase = ti * 16 + ((lane >> 4) << 3);
#pragma unroll
    for (int v = 0; v < 8; ++v) {
        int row = rbase + v;
        float val = c[v];
        val = val > 0.0f ? val : 0.0f;
        if (row == col) val = 0.0f;
        Ab[(size_t)row * NB + col] = (_Float16)val;
    }
}

// ---------------- 4) invdeg = 1/max(rowsum(A),1e-6) ----------------
__global__ __launch_bounds__(256) void deg_kernel(const _Float16* __restrict__ A,
                                                  float* __restrict__ invdeg) {
    int row = blockIdx.x, t = threadIdx.x;     // row = b*NB + n (A contiguous)
    const _Float16* ar = A + (size_t)row * NB;
    float s = 0.0f;
    for (int i = t; i < NB; i += 256) s += (float)ar[i];
    __shared__ float red[256];
    red[t] = s;  __syncthreads();
    for (int st = 128; st > 0; st >>= 1) { if (t < st) red[t] += red[t + st]; __syncthreads(); }
    if (t == 0) { float d = red[0] > 1e-6f ? red[0] : 1e-6f; invdeg[row] = 1.0f / d; }
}

// ---------------- 5a) neigh = (A @ x) * invdeg : 4 waves/block (one per 16-col tile) --------
// B operand fed from XT (features pre-transposed) so every lane load is contiguous.
__global__ __launch_bounds__(128) void aggregate_kernel(const _Float16* __restrict__ A,
                                                        const _Float16* __restrict__ XT,
                                                        const float* __restrict__ invdeg,
                                                        float* __restrict__ neigh) {
    int ti = blockIdx.x, b = blockIdx.z;
    int tj = threadIdx.y;                       // 0..3
    int lane = threadIdx.x;
    const _Float16* Ab  = A  + (size_t)b * NB * NB + (size_t)ti * 16 * NB;
    const _Float16* XTb = XT + (size_t)b * DIM * NB + (size_t)tj * 16 * NB;
    int r = lane & 15;
    v8f c = {};
    for (int kt = 0; kt < NB / 32; ++kt) {
        int pk = (kt + 8 < NB / 32) ? kt + 8 : kt;           // clamped, no divergence
        __builtin_prefetch(Ab + (size_t)r * NB + pk * 32, 0, 0);
        v16h af = frag_rowmajor(Ab + kt * 32, NB, lane);
        v16h bf = frag_rowmajor(XTb + kt * 32, NB, lane);
        c = __builtin_amdgcn_wmma_f32_16x16x32_f16(false, af, false, bf, (short)0, c, false, false);
    }
    int col   = tj * 16 + (lane & 15);
    int rbase = ti * 16 + ((lane >> 4) << 3);
#pragma unroll
    for (int v = 0; v < 8; ++v) {
        int row = rbase + v;
        neigh[((size_t)b * NB + row) * DIM + col] = c[v] * invdeg[b * NB + row];
    }
}

// ---------------- 5b) x' = x@Wself^T + bs + neigh@Wneigh^T + bn ----------------
__global__ __launch_bounds__(64) void gnn_linear_kernel(
    const float* __restrict__ X, const float* __restrict__ neigh,
    const float* __restrict__ Ws, const float* __restrict__ Bs,
    const float* __restrict__ Wn, const float* __restrict__ Bn,
    float* __restrict__ Xout)
{
    int row = blockIdx.x, t = threadIdx.x;
    __shared__ float xs[64], ns[64];
    xs[t] = X[(size_t)row * DIM + t];
    ns[t] = neigh[(size_t)row * DIM + t];
    __syncthreads();
    float acc = Bs[t] + Bn[t];
    const float* ws = Ws + t * DIM;
    const float* wn = Wn + t * DIM;
#pragma unroll 8
    for (int i = 0; i < DIM; ++i) acc += xs[i] * ws[i] + ns[i] * wn[i];
    Xout[(size_t)row * DIM + t] = acc;
}

// ---------------- 6) LayerNorm + attention score ----------------
__global__ __launch_bounds__(64) void ln_attn_kernel(
    const float* __restrict__ X, const float* __restrict__ lng, const float* __restrict__ lnb,
    const float* __restrict__ aw1, const float* __restrict__ ab1,
    const float* __restrict__ aw2, const float* __restrict__ ab2,
    const unsigned char* __restrict__ mask,
    float* __restrict__ Xln, float* __restrict__ scores)
{
    int row = blockIdx.x, t = threadIdx.x;
    __shared__ float xs[64], red[64];
    float v = X[(size_t)row * DIM + t];
    red[t] = v;  __syncthreads();
    for (int s = 32; s > 0; s >>= 1) { if (t < s) red[t] += red[t + s]; __syncthreads(); }
    float mu = red[0] * (1.0f / 64.0f);
    __syncthreads();
    float dv = v - mu;
    red[t] = dv * dv;  __syncthreads();
    for (int s = 32; s > 0; s >>= 1) { if (t < s) red[t] += red[t + s]; __syncthreads(); }
    float var = red[0] * (1.0f / 64.0f);
    __syncthreads();
    float xl = dv * rsqrtf(var + 1e-5f) * lng[t] + lnb[t];
    xs[t] = xl;
    Xln[(size_t)row * DIM + t] = xl;
    __syncthreads();
    float h = ab1[t];
    const float* w = aw1 + t * DIM;
#pragma unroll 8
    for (int i = 0; i < DIM; ++i) h += xs[i] * w[i];
    h = tanhf(h);
    red[t] = h * aw2[t];  __syncthreads();
    for (int s = 32; s > 0; s >>= 1) { if (t < s) red[t] += red[t + s]; __syncthreads(); }
    if (t == 0) {
        float s = red[0] + ab2[0];
        scores[row] = mask[row] ? s : -1e9f;
    }
}

// ---------------- 7) softmax over N + weighted pool ----------------
__global__ __launch_bounds__(256) void pool_kernel(const float* __restrict__ Xln,
                                                   const float* __restrict__ scores,
                                                   float* __restrict__ pooled) {
    int b = blockIdx.x, t = threadIdx.x;
    __shared__ float w[NB];
    __shared__ float red[256];
    const float* sc = scores + (size_t)b * NB;
    float mx = -1e30f;
    for (int i = t; i < NB; i += 256) mx = fmaxf(mx, sc[i]);
    red[t] = mx;  __syncthreads();
    for (int s = 128; s > 0; s >>= 1) { if (t < s) red[t] = fmaxf(red[t], red[t + s]); __syncthreads(); }
    mx = red[0];  __syncthreads();
    float sum = 0.0f;
    for (int i = t; i < NB; i += 256) { float e = __expf(sc[i] - mx); w[i] = e; sum += e; }
    red[t] = sum;  __syncthreads();
    for (int s = 128; s > 0; s >>= 1) { if (t < s) red[t] += red[t + s]; __syncthreads(); }
    float inv = 1.0f / red[0];
    __syncthreads();
    int d = t & 63, chunk = t >> 6;                 // 4 chunks of 256 rows
    const float* xb = Xln + (size_t)b * NB * DIM;
    float acc = 0.0f;
    for (int n = chunk * 256; n < (chunk + 1) * 256; ++n) acc += w[n] * xb[(size_t)n * DIM + d];
    red[t] = acc;  __syncthreads();
    if (t < 64) pooled[b * DIM + d] = (red[d] + red[d + 64] + red[d + 128] + red[d + 192]) * inv;
}

// ---------------- 8) classifier ----------------
__global__ __launch_bounds__(128) void cls_kernel(const float* __restrict__ pooled,
                                                  const float* __restrict__ w1, const float* __restrict__ b1,
                                                  const float* __restrict__ w2, const float* __restrict__ b2,
                                                  float* __restrict__ out) {
    int b = blockIdx.x, t = threadIdx.x;
    __shared__ float ps[64], hs[128];
    if (t < 64) ps[t] = pooled[b * DIM + t];
    __syncthreads();
    float h = b1[t];
    const float* w = w1 + t * DIM;
#pragma unroll 8
    for (int i = 0; i < DIM; ++i) h += ps[i] * w[i];
    hs[t] = h > 0.0f ? h : 0.0f;
    __syncthreads();
    if (t < 4) {
        float acc = b2[t];
        const float* wr = w2 + t * 128;
#pragma unroll 8
        for (int i = 0; i < 128; ++i) acc += hs[i] * wr[i];
        out[b * 4 + t] = acc;
    }
}

// ---------------- host launch ----------------
extern "C" void kernel_launch(void* const* d_in, const int* in_sizes, int n_in,
                              void* d_out, int out_size, void* d_ws, size_t ws_size,
                              hipStream_t stream) {
    const float* beats   = (const float*)d_in[0];
    const float* rr      = (const float*)d_in[1];
    const unsigned char* mask = (const unsigned char*)d_in[2];
    const float* conv1_w = (const float*)d_in[3];
    const float* conv1_b = (const float*)d_in[4];
    const float* conv2_w = (const float*)d_in[5];
    const float* conv2_b = (const float*)d_in[6];
    const float* fc_w    = (const float*)d_in[7];
    const float* fc_b    = (const float*)d_in[8];
    const float* self_w  = (const float*)d_in[9];
    const float* self_b  = (const float*)d_in[10];
    const float* neigh_w = (const float*)d_in[11];
    const float* neigh_b = (const float*)d_in[12];
    const float* attn_w1 = (const float*)d_in[13];
    const float* attn_b1 = (const float*)d_in[14];
    const float* attn_w2 = (const float*)d_in[15];
    const float* attn_b2 = (const float*)d_in[16];
    const float* ln_g    = (const float*)d_in[17];
    const float* ln_b    = (const float*)d_in[18];
    const float* cls_w1  = (const float*)d_in[19];
    const float* cls_b1  = (const float*)d_in[20];
    const float* cls_w2  = (const float*)d_in[21];
    const float* cls_b2  = (const float*)d_in[22];
    float* out = (float*)d_out;

    char* ws = (char*)d_ws;
    const size_t MB = 1u << 20;
    float*    X0     = (float*)(ws + 0 * MB);            // 4 MB  (BN x 64 f32)
    float*    X1     = (float*)(ws + 4 * MB);            // 4 MB
    _Float16* Xn     = (_Float16*)(ws + 8 * MB);         // 2 MB
    _Float16* XT     = (_Float16*)(ws + 10 * MB);        // 2 MB  (B x 64 x NB f16)
    float*    neigh  = (float*)(ws + 12 * MB);           // 4 MB
    float*    invdeg = (float*)(ws + 16 * MB);           // 64 KB
    float*    scores = (float*)(ws + 16 * MB + 65536);   // 64 KB
    float*    pooled = (float*)(ws + 16 * MB + 2 * 65536);
    _Float16* w2r    = (_Float16*)(ws + 16 * MB + 3 * 65536); // 20 KB
    _Float16* A      = (_Float16*)(ws + 17 * MB);        // 32 MB (B x N x N f16)

    convert_w2_kernel<<<40, 256, 0, stream>>>(conv2_w, w2r);
    encoder_kernel<<<BN, 256, 0, stream>>>(beats, rr, conv1_w, conv1_b, conv2_b,
                                           w2r, fc_w, fc_b, X0);
    rownorm_kernel<<<BN, 64, 0, stream>>>(X0, Xn);
    adjacency_kernel<<<dim3(NB / 16, NB / 128, BATCH), dim3(32, 8), 0, stream>>>(Xn, A);
    deg_kernel<<<BN, 256, 0, stream>>>(A, invdeg);

    float* Xcur = X0;
    float* Xnxt = X1;
    for (int l = 0; l < 2; ++l) {
        transpose16_kernel<<<(BN * DIM + 255) / 256, 256, 0, stream>>>(Xcur, XT);
        aggregate_kernel<<<dim3(NB / 16, 1, BATCH), dim3(32, 4), 0, stream>>>(A, XT, invdeg, neigh);
        gnn_linear_kernel<<<BN, 64, 0, stream>>>(Xcur, neigh,
                                                 self_w + l * DIM * DIM, self_b + l * DIM,
                                                 neigh_w + l * DIM * DIM, neigh_b + l * DIM, Xnxt);
        float* tmp = Xcur; Xcur = Xnxt; Xnxt = tmp;
    }
    // Xcur == X0 after 2 layers; write LayerNorm'd features into X1
    ln_attn_kernel<<<BN, 64, 0, stream>>>(Xcur, ln_g, ln_b, attn_w1, attn_b1,
                                          attn_w2, attn_b2, mask, Xnxt, scores);
    pool_kernel<<<BATCH, 256, 0, stream>>>(Xnxt, scores, pooled);
    cls_kernel<<<BATCH, 128, 0, stream>>>(pooled, cls_w1, cls_b1, cls_w2, cls_b2, out);
}
